// LengthRegulator_34059090657417
// MI455X (gfx1250) — matrix operand order repaired and checked by reading, not compiled, not run on hardware
//
#include <hip/hip_runtime.h>
#include <hip/hip_bf16.h>

typedef __attribute__((ext_vector_type(16))) _Float16 v16h;
typedef __attribute__((ext_vector_type(8)))  _Float16 v8h;
typedef __attribute__((ext_vector_type(8)))  float    v8f;

#define B_   8
#define T_   512
#define D_   384
#define K_   3
#define L_   4096
#define KTOT 1152      // 3*384 (im2col K)
#define NKT  36        // 1152 / 32
#define NNT  24        // 384 / 16
#define M1   510       // conv1 valid rows per batch
#define M2   508       // conv2 valid rows per batch
#define PACKN (NNT*NKT*32*16)   // 442368 packed halfs per conv layer

// ---------------- weight repack: [Cout,Cin,3] f32 -> WMMA B-fragment f16 ----
// b-frag element i of lane within tile (nt,kt):
//   N = nt*16 + lane%16 ;  Kglob = kt*32 + (i/8)*16 + (lane/16)*8 + i%8
//   W'[Kglob, N] = w[co=N][ci=Kglob%384][kk=Kglob/384]
__global__ void pack_w_kernel(const float* __restrict__ w, _Float16* __restrict__ wp) {
    int p = blockIdx.x * 256 + threadIdx.x;
    if (p >= PACKN) return;
    int i    =  p        & 15;
    int lane = (p >> 4)  & 31;
    int kt   = (p >> 9)  % NKT;
    int nt   = (p >> 9)  / NKT;
    int n  = nt * 16 + (lane & 15);
    int kg = kt * 32 + ((i >> 3) << 4) + ((lane >> 4) << 3) + (i & 7);
    int ci = kg % D_;
    int kk = kg / D_;
    wp[p] = (_Float16)w[(n * D_ + ci) * K_ + kk];
}

// ---------------- fp32 -> fp16 activation copy ------------------------------
__global__ void to_f16_kernel(const float* __restrict__ x, _Float16* __restrict__ xh, int n) {
    int p = blockIdx.x * 256 + threadIdx.x;
    if (p < n) xh[p] = (_Float16)x[p];
}

// ---------------- per-batch cumsum of integer durations ---------------------
__global__ void cumsum_kernel(const int* __restrict__ tgt, int* __restrict__ cum) {
    int b = threadIdx.x;
    if (b >= B_) return;
    int run = 0;
    for (int t = 0; t < T_; ++t) { run += tgt[b * T_ + t]; cum[b * T_ + t] = run; }
}

// ---------------- fused conv1d(K=3,valid) + bias + LN + ReLU [+ lin+exp] ----
// grid = B_*32 blocks (one 16-row M-tile per block), 256 threads = 8 waves,
// wave w computes N-tiles 3w..3w+2 with v_wmma_f32_16x16x32_f16.
template<bool SECOND>
__global__ __launch_bounds__(256) void conv_ln_kernel(
    const _Float16* __restrict__ src,   // [B_*T_, 384] f16 activations
    const _Float16* __restrict__ wp,    // packed B fragments
    const float* __restrict__ cb,       // conv bias [384]
    const float* __restrict__ g,        // ln gamma [384]
    const float* __restrict__ be,       // ln beta  [384]
    _Float16* __restrict__ hout,        // layer1: f16 output [B_*T_,384]
    const float* __restrict__ lw,       // layer2: lin_w [384]
    const float* __restrict__ lb,       // layer2: lin_b [1]
    float* __restrict__ dur)            // layer2: durations [B_,508]
{
    const int Mb   = SECOND ? M2 : M1;
    const int blk  = blockIdx.x;
    const int b    = blk >> 5;
    const int mt   = blk & 31;
    const int tid  = threadIdx.x;
    const int lane = tid & 31;
    const int wv   = tid >> 5;

    __shared__ float s_tile[16][388];       // 16 rows x 384 cols (+pad)
    __shared__ float s_r1[16][16];
    __shared__ float s_r2[16][16];
    __shared__ float s_mean[16], s_rstd[16];

    // A fragment source row (clamped for tail tiles -> memory safe)
    const int mrow = lane & 15;
    const int r    = mt * 16 + mrow;
    const int rc   = (r < Mb) ? r : (Mb - 1);
    const _Float16* arow = src + (size_t)(b * T_ + rc) * D_;   // 1152 contiguous halfs
    const int khi = (lane >> 4) << 3;                          // K sub-offset 0 / 8

    v8f c0 = {}, c1 = {}, c2 = {};
    const int nt0 = wv * 3;

    for (int kt = 0; kt < NKT; ++kt) {
        const int k0 = kt * 32;
        union { v16h v; struct { v8h lo, hi; } s; } ua;
        ua.s.lo = *(const v8h*)(arow + k0 + khi);        // K = k0 + khi + 0..7
        ua.s.hi = *(const v8h*)(arow + k0 + 16 + khi);   // K = k0 + 16 + khi + 0..7
        const v16h a = ua.v;

        const v16h b0 = *(const v16h*)(wp + (((size_t)(nt0 + 0) * NKT + kt) * 32 + lane) * 16);
        const v16h b1 = *(const v16h*)(wp + (((size_t)(nt0 + 1) * NKT + kt) * 32 + lane) * 16);
        const v16h b2 = *(const v16h*)(wp + (((size_t)(nt0 + 2) * NKT + kt) * 32 + lane) * 16);

        c0 = __builtin_amdgcn_wmma_f32_16x16x32_f16(false, a, false, b0, (short)0, c0, false, false);
        c1 = __builtin_amdgcn_wmma_f32_16x16x32_f16(false, a, false, b1, (short)0, c1, false, false);
        c2 = __builtin_amdgcn_wmma_f32_16x16x32_f16(false, a, false, b2, (short)0, c2, false, false);
    }

    // spill C tiles (+conv bias) to LDS: lane holds col = L%16, rows v + (L/16)*8
    const int colL   = lane & 15;
    const int rowoff = (lane >> 4) << 3;
    {
        int col = (nt0 + 0) * 16 + colL; float bv = cb[col];
        #pragma unroll
        for (int v = 0; v < 8; ++v) s_tile[rowoff + v][col] = c0[v] + bv;
    }
    {
        int col = (nt0 + 1) * 16 + colL; float bv = cb[col];
        #pragma unroll
        for (int v = 0; v < 8; ++v) s_tile[rowoff + v][col] = c1[v] + bv;
    }
    {
        int col = (nt0 + 2) * 16 + colL; float bv = cb[col];
        #pragma unroll
        for (int v = 0; v < 8; ++v) s_tile[rowoff + v][col] = c2[v] + bv;
    }
    __syncthreads();

    // LayerNorm: mean / population-variance over 384 cols per row
    const int row = tid >> 4, sub = tid & 15;
    float s = 0.f, s2 = 0.f;
    #pragma unroll
    for (int j = 0; j < NNT; ++j) {
        float v = s_tile[row][sub + j * 16];
        s += v; s2 += v * v;
    }
    s_r1[row][sub] = s; s_r2[row][sub] = s2;
    __syncthreads();
    if (tid < 16) {
        float ts = 0.f, ts2 = 0.f;
        #pragma unroll
        for (int j = 0; j < 16; ++j) { ts += s_r1[tid][j]; ts2 += s_r2[tid][j]; }
        float m   = ts  * (1.f / 384.f);
        float var = ts2 * (1.f / 384.f) - m * m;
        s_mean[tid] = m;
        s_rstd[tid] = rsqrtf(var + 1e-5f);
    }
    __syncthreads();

    const float m  = s_mean[row];
    const float rs = s_rstd[row];
    const bool valid = (mt * 16 + row) < Mb;
    float dot = 0.f;
    #pragma unroll
    for (int j = 0; j < NNT; ++j) {
        int col = sub + j * 16;
        float v = (s_tile[row][col] - m) * rs * g[col] + be[col];
        v = v > 0.f ? v : 0.f;                              // ReLU
        if (SECOND) {
            dot += v * lw[col];
        } else if (valid) {
            hout[(size_t)(b * T_ + mt * 16 + row) * D_ + col] = (_Float16)v;
        }
    }
    if (SECOND) {
        __syncthreads();
        s_r1[row][sub] = dot;
        __syncthreads();
        if (tid < 16) {
            float td = 0.f;
            #pragma unroll
            for (int j = 0; j < 16; ++j) td += s_r1[tid][j];
            td += lb[0];
            td = td > 0.f ? td : 0.f;                       // ReLU
            int rr = mt * 16 + tid;
            if (rr < M2) dur[b * M2 + rr] = expf(td);       // exp(durations)
        }
    }
}

// ---------------- length regulation: binary-search gather -------------------
// block = 256 thr handles 8 output positions l of batch b; row copy as float4.
__global__ __launch_bounds__(256) void gather_kernel(
    const float* __restrict__ x, const int* __restrict__ cum, float* __restrict__ out)
{
    __shared__ int s_cum[T_];
    const int blk   = blockIdx.x;
    const int b     = blk >> 9;            // 512 blocks per batch
    const int lbase = (blk & 511) * 8;
    const int tid   = threadIdx.x;
    s_cum[tid]       = cum[b * T_ + tid];
    s_cum[tid + 256] = cum[b * T_ + tid + 256];
    __syncthreads();

    const int lane = tid & 31;
    const int l    = lbase + (tid >> 5);
    // smallest t with cum[t] > l  (zero-duration tokens never selected)
    int lo = 0, hi = T_;
    while (lo < hi) { int mid = (lo + hi) >> 1; if (s_cum[mid] <= l) lo = mid + 1; else hi = mid; }

    float4* orow = (float4*)(out + (size_t)(b * L_ + l) * D_);
    if (lo < T_) {
        const float4* xrow = (const float4*)(x + (size_t)(b * T_ + lo) * D_);
        #pragma unroll
        for (int j = 0; j < 3; ++j) orow[lane + 32 * j] = xrow[lane + 32 * j];
    } else {
        const float4 z = {0.f, 0.f, 0.f, 0.f};
        #pragma unroll
        for (int j = 0; j < 3; ++j) orow[lane + 32 * j] = z;
    }
}

// ---------------------------------------------------------------------------
extern "C" void kernel_launch(void* const* d_in, const int* in_sizes, int n_in,
                              void* d_out, int out_size, void* d_ws, size_t ws_size,
                              hipStream_t stream) {
    const float* x   = (const float*)d_in[0];
    const int*   tgt = (const int*)d_in[1];
    /* d_in[2] = max_len (fixed 4096) */
    const float* c1w = (const float*)d_in[3];
    const float* c1b = (const float*)d_in[4];
    const float* g1  = (const float*)d_in[5];
    const float* b1  = (const float*)d_in[6];
    const float* c2w = (const float*)d_in[7];
    const float* c2b = (const float*)d_in[8];
    const float* g2  = (const float*)d_in[9];
    const float* b2  = (const float*)d_in[10];
    const float* lw  = (const float*)d_in[11];
    const float* lb  = (const float*)d_in[12];

    // workspace carve (256B aligned sections)
    char* ws = (char*)d_ws;
    size_t off = 0;
    auto carve = [&](size_t bytes) -> void* {
        off = (off + 255) & ~(size_t)255;
        void* p = ws + off;
        off += bytes;
        return p;
    };
    int*      cum = (int*)     carve((size_t)B_ * T_ * sizeof(int));
    _Float16* xh  = (_Float16*)carve((size_t)B_ * T_ * D_ * sizeof(_Float16));
    _Float16* h1h = (_Float16*)carve((size_t)B_ * T_ * D_ * sizeof(_Float16));
    _Float16* w1p = (_Float16*)carve((size_t)PACKN * sizeof(_Float16));
    _Float16* w2p = (_Float16*)carve((size_t)PACKN * sizeof(_Float16));

    float* out_main = (float*)d_out;                              // [8,4096,384]
    float* dur      = out_main + (size_t)B_ * L_ * D_;            // [8,508]

    pack_w_kernel<<<(PACKN + 255) / 256, 256, 0, stream>>>(c1w, w1p);
    pack_w_kernel<<<(PACKN + 255) / 256, 256, 0, stream>>>(c2w, w2p);
    to_f16_kernel<<<(B_ * T_ * D_ + 255) / 256, 256, 0, stream>>>(x, xh, B_ * T_ * D_);
    cumsum_kernel<<<1, 32, 0, stream>>>(tgt, cum);

    conv_ln_kernel<false><<<B_ * 32, 256, 0, stream>>>(
        xh, w1p, c1b, g1, b1, h1h, nullptr, nullptr, nullptr);
    conv_ln_kernel<true><<<B_ * 32, 256, 0, stream>>>(
        h1h, w2p, c2b, g2, b2, nullptr, lw, lb, dur);

    gather_kernel<<<B_ * 512, 256, 0, stream>>>(x, cum, out_main);
}
